// GATv2Encoder_37915971289664
// MI455X (gfx1250) — matrix operand order repaired and checked by reading, not compiled, not run on hardware
//
#include <hip/hip_runtime.h>
#include <hip/hip_bf16.h>

// ---------------------------------------------------------------------------
// GATv2 encoder for MI455X (gfx1250): bf16 WMMA GEMMs + edge scatter kernels.
// Round 3: software-pipelined B fragments + hoisted bias epilogue.
// ---------------------------------------------------------------------------

typedef __attribute__((ext_vector_type(16))) __bf16 v16bf;
typedef __attribute__((ext_vector_type(8)))  float  v8f;
typedef __attribute__((ext_vector_type(8)))  unsigned int v8u;

union Frag16 { v8u u; v16bf bf; };

#define NN    50000
#define EE    800000
#define INF_C 256
#define HIDC  128
#define OUTC  128
#define HEADS1 4
#define EDC   64
#define NEG_SLOPE 0.2f

#define TM 64        // rows per block
#define TK 32        // K per WMMA step (bf16)
#define NCOLS 128    // every GEMM here has 128 output columns
#define APAD 8       // pad A rows to 40 bf16 = 80 B (16B-aligned, bank-spread)
#define BPAD 8       // pad Bt rows to 40 bf16 = 80 B

__device__ __forceinline__ Frag16 load_frag_2x16B(const uint4* p0, const uint4* p1) {
    uint4 a = *p0, b = *p1;
    Frag16 f;
    f.u[0] = a.x; f.u[1] = a.y; f.u[2] = a.z; f.u[3] = a.w;
    f.u[4] = b.x; f.u[5] = b.y; f.u[6] = b.z; f.u[7] = b.w;
    return f;
}

// ------------------------- generic WMMA GEMM -------------------------------
// C[M x 128] = act(A[M x K] @ W[K x 128] + bias), f32 in/out, bf16 WMMA core.
// act: 0 = none, 1 = ELU
__global__ __launch_bounds__(128) void gemm_bias_act_wmma(
    const float* __restrict__ A, const float* __restrict__ W,
    const float* __restrict__ bias, float* __restrict__ C,
    int M, int K, int act)
{
    __shared__ __bf16 sA[TM][TK + APAD];      // 64 x 40 bf16 = 5 KB
    __shared__ __bf16 sBt[NCOLS][TK + BPAD];  // 128 x 40 bf16 = 10 KB (transposed)

    const int tid  = threadIdx.x;       // 0..127
    const int wave = tid >> 5;          // 0..3
    const int lane = tid & 31;
    const int hi   = lane >> 4;         // half-wave selector
    const int lm   = lane & 15;
    const long row0 = (long)blockIdx.x * TM;

    v8f acc[8] = {};

    for (int k0 = 0; k0 < K; k0 += TK) {
        // ---- stage A tile (64 x 32) as bf16, float4 global loads ----
        for (int i = tid; i < (TM * TK) / 4; i += 128) {
            int r  = i >> 3;          // /8 float4 per row
            int c4 = i & 7;
            long row = row0 + r;
            float4 f = make_float4(0.f, 0.f, 0.f, 0.f);
            if (row < M)
                f = *(const float4*)(A + row * (long)K + k0 + 4 * c4);
            __bf16 t[4] = { (__bf16)f.x, (__bf16)f.y, (__bf16)f.z, (__bf16)f.w };
            *(uint2*)(&sA[r][4 * c4]) = *(const uint2*)t;   // ds_store_b64
        }
        // ---- stage W tile (32 x 128) TRANSPOSED as bf16 ----
        for (int i = tid; i < (TK * NCOLS) / 4; i += 128) {
            int r  = i >> 5;          // /32 float4 per row
            int c4 = i & 31;
            float4 f = *(const float4*)(W + (long)(k0 + r) * NCOLS + 4 * c4);
            sBt[4 * c4 + 0][r] = (__bf16)f.x;
            sBt[4 * c4 + 1][r] = (__bf16)f.y;
            sBt[4 * c4 + 2][r] = (__bf16)f.z;
            sBt[4 * c4 + 3][r] = (__bf16)f.w;
        }
        __syncthreads();

        // ---- A fragment: two contiguous 16B chunks (ISA 7.12.2 layout) ----
        const uint4* rowA = (const uint4*)(&sA[16 * wave + lm][0]);  // 80B row
        Frag16 af = load_frag_2x16B(&rowA[hi], &rowA[2 + hi]);

        // ---- software-pipelined B fragments: column nt*16+lm, K chunks hi and 2+hi (same as A) ----
        const uint4* rowB0 = (const uint4*)(&sBt[lm][0]);
        Frag16 bcur = load_frag_2x16B(&rowB0[hi], &rowB0[2 + hi]);
#pragma unroll
        for (int nt = 0; nt < 8; ++nt) {
            Frag16 bnext;
            if (nt < 7) {
                const uint4* rowB = (const uint4*)(&sBt[(nt + 1) * 16 + lm][0]);
                bnext = load_frag_2x16B(&rowB[hi], &rowB[2 + hi]);
            }
            acc[nt] = __builtin_amdgcn_wmma_f32_16x16x32_bf16(
                false, af.bf, false, bcur.bf, (short)0, acc[nt], false, false);
            if (nt < 7) bcur = bnext;
        }
        __syncthreads();
    }

    // ---- epilogue: C/D layout VGPR r -> row = r + 8*hi, col = lane%16 ----
#pragma unroll
    for (int nt = 0; nt < 8; ++nt) {
        const int col = nt * 16 + lm;
        const float bv = (bias != nullptr) ? bias[col] : 0.0f;  // hoisted, uniform branch
#pragma unroll
        for (int r = 0; r < 8; ++r) {
            long row = row0 + 16 * wave + r + 8 * hi;
            if (row < M) {
                float v = acc[nt][r] + bv;
                if (act == 1) v = (v > 0.0f) ? v : (__expf(v) - 1.0f);
                C[row * (long)NCOLS + col] = v;
            }
        }
    }
}

// ----------------------- monotone float<->uint encoding --------------------
__device__ __forceinline__ unsigned enc_f32(float f) {
    unsigned b = __float_as_uint(f);
    return (b & 0x80000000u) ? ~b : (b | 0x80000000u);
}
__device__ __forceinline__ float dec_f32(unsigned u) {
    return __uint_as_float((u & 0x80000000u) ? (u ^ 0x80000000u) : ~u);
}

// --------------------------- self-loop edge attrs --------------------------
__global__ void count_edges_k(const int* __restrict__ dst, float* __restrict__ cnt, int E) {
    int e = blockIdx.x * blockDim.x + threadIdx.x;
    if (e < E) atomicAdd(&cnt[dst[e]], 1.0f);
}

__global__ void sum_loop_ea_k(const float* __restrict__ ea, const int* __restrict__ dst,
                              float* __restrict__ loop_sum, int E) {
    int widx = (blockIdx.x * blockDim.x + threadIdx.x) >> 5;
    int lane = threadIdx.x & 31;
    if (widx >= E) return;
    int d = dst[widx];
    const float* row = ea + (size_t)widx * HIDC;
    float* out = loop_sum + (size_t)d * HIDC;
    for (int c = lane; c < HIDC; c += 32) atomicAdd(&out[c], row[c]);
}

__global__ void finalize_loop_k(const float* __restrict__ loop_sum, const float* __restrict__ cnt,
                                float* __restrict__ ea_loop_rows, int n_total) {
    long i = (long)blockIdx.x * blockDim.x + threadIdx.x;
    if (i >= (long)n_total * HIDC) return;
    int n = (int)(i >> 7);
    float ct = cnt[n];
    float v = (ct > 0.0f) ? (loop_sum[i] / fmaxf(ct, 1.0f)) : 0.0f;
    ea_loop_rows[i] = v;
}

// ----------------------------- attention kernels ---------------------------
// one wave per (edge or self-loop); lanes stride channels; shfl reduction
__global__ void edge_logits_k(const float* __restrict__ xl, const float* __restrict__ xr,
                              const float* __restrict__ ew, const float* __restrict__ att,
                              const int* __restrict__ src, const int* __restrict__ dst,
                              float* __restrict__ logit, unsigned* __restrict__ amax,
                              int E, int Nn, int heads, int C) {
    int widx = (blockIdx.x * blockDim.x + threadIdx.x) >> 5;
    int lane = threadIdx.x & 31;
    if (widx >= E + Nn) return;
    int s, d;
    if (widx < E) { s = src[widx]; d = dst[widx]; } else { s = d = widx - E; }
    const int HC = heads * C;
    const float* er = ew + (size_t)widx * HC;
    const float* xs = xl + (size_t)s * HC;
    const float* xd = xr + (size_t)d * HC;
    for (int h = 0; h < heads; ++h) {
        float acc = 0.0f;
        for (int c = lane; c < C; c += 32) {
            int k = h * C + c;
            float m = xs[k] + xd[k] + er[k];
            m = (m > 0.0f) ? m : NEG_SLOPE * m;
            acc += m * att[k];
        }
#pragma unroll
        for (int off = 16; off > 0; off >>= 1) acc += __shfl_xor(acc, off, 32);
        if (lane == 0) {
            logit[(size_t)widx * heads + h] = acc;
            atomicMax(&amax[(size_t)d * heads + h], enc_f32(acc));
        }
    }
}

// alpha = exp(logit - amax[dst]); denom[dst] += alpha    (in-place on logit buf)
__global__ void edge_softmax_k(float* __restrict__ logit_alpha, const unsigned* __restrict__ amax,
                               float* __restrict__ denom, const int* __restrict__ dst,
                               int E, int Nn, int heads) {
    long i = (long)blockIdx.x * blockDim.x + threadIdx.x;
    long total = (long)(E + Nn) * heads;
    if (i >= total) return;
    int ef = (int)(i / heads);
    int h  = (int)(i - (long)ef * heads);
    int d  = (ef < E) ? dst[ef] : (ef - E);
    float a = __expf(logit_alpha[i] - dec_f32(amax[(size_t)d * heads + h]));
    logit_alpha[i] = a;
    atomicAdd(&denom[(size_t)d * heads + h], a);
}

// agg[dst] += (alpha/denom[dst]) * xl[src]   (one wave per edge)
__global__ void edge_aggregate_k(const float* __restrict__ alpha, const float* __restrict__ denom,
                                 const float* __restrict__ xl, const int* __restrict__ src,
                                 const int* __restrict__ dst, float* __restrict__ agg,
                                 int E, int Nn, int heads, int C) {
    int widx = (blockIdx.x * blockDim.x + threadIdx.x) >> 5;
    int lane = threadIdx.x & 31;
    if (widx >= E + Nn) return;
    int s, d;
    if (widx < E) { s = src[widx]; d = dst[widx]; } else { s = d = widx - E; }
    const int HC = heads * C;
    const float* xs = xl + (size_t)s * HC;
    float* ag = agg + (size_t)d * HC;
    for (int h = 0; h < heads; ++h) {
        float w = alpha[(size_t)widx * heads + h] / denom[(size_t)d * heads + h];
        for (int c = lane; c < C; c += 32)
            atomicAdd(&ag[h * C + c], w * xs[h * C + c]);
    }
}

// out = act(agg + bias)
__global__ void finalize_layer_k(const float* __restrict__ agg, const float* __restrict__ bias,
                                 float* __restrict__ out, int act, int n_total) {
    long i = (long)blockIdx.x * blockDim.x + threadIdx.x;
    if (i >= (long)n_total * HIDC) return;
    float v = agg[i] + bias[(int)(i & 127)];
    if (act == 1) v = (v > 0.0f) ? v : (__expf(v) - 1.0f);
    out[i] = v;
}

// ------------------------------- host side ---------------------------------
static inline int cdiv(long a, long b) { return (int)((a + b - 1) / b); }

extern "C" void kernel_launch(void* const* d_in, const int* in_sizes, int n_in,
                              void* d_out, int out_size, void* d_ws, size_t ws_size,
                              hipStream_t stream) {
    (void)in_sizes; (void)n_in; (void)out_size; (void)ws_size;

    const float* x         = (const float*)d_in[0];
    const int*   src       = (const int*)d_in[1];
    const int*   dst       = src + EE;
    const float* edge_attr = (const float*)d_in[2];
    const float* W_in = (const float*)d_in[3];  const float* b_in = (const float*)d_in[4];
    const float* W_ep = (const float*)d_in[5];  const float* b_ep = (const float*)d_in[6];
    const float* Wl1  = (const float*)d_in[7];  const float* bl1  = (const float*)d_in[8];
    const float* Wr1  = (const float*)d_in[9];  const float* br1  = (const float*)d_in[10];
    const float* We1  = (const float*)d_in[11]; const float* att1 = (const float*)d_in[12];
    const float* bias1= (const float*)d_in[13];
    const float* Wl2  = (const float*)d_in[14]; const float* bl2  = (const float*)d_in[15];
    const float* Wr2  = (const float*)d_in[16]; const float* br2  = (const float*)d_in[17];
    const float* We2  = (const float*)d_in[18]; const float* att2 = (const float*)d_in[19];
    const float* bias2= (const float*)d_in[20];
    float* z_out = (float*)d_out;

    const long EN = (long)EE + NN;     // edges incl. self loops
    const size_t NH = (size_t)NN * HIDC;
    const size_t ENH = (size_t)EN * HIDC;

    // workspace carve-out (floats)
    float* ws       = (float*)d_ws;
    float* h        = ws;                         // N x 128
    float* ea_f     = h + NH;                     // (E+N) x 128
    float* xl       = ea_f + ENH;                 // N x 128
    float* xr       = xl + NH;                    // N x 128
    float* ew       = xr + NH;                    // (E+N) x 128
    float* alpha    = ew + ENH;                   // (E+N) x HEADS1
    unsigned* amax  = (unsigned*)(alpha + (size_t)EN * HEADS1);  // N x HEADS1
    float* denom    = (float*)(amax + (size_t)NN * HEADS1);      // N x HEADS1
    float* cnt      = denom + (size_t)NN * HEADS1;               // N
    float* loop_sum = cnt + NN;                   // N x 128
    float* agg      = loop_sum + NH;              // N x 128
    float* h1       = agg + NH;                   // N x 128

    dim3 blk128(128), blk256(256);

    // 1) h = elu(x @ W_in + b_in)
    gemm_bias_act_wmma<<<cdiv(NN, TM), blk128, 0, stream>>>(x, W_in, b_in, h, NN, INF_C, 1);
    // 2) ea_f[0:E] = elu(edge_attr @ W_ep + b_ep)
    gemm_bias_act_wmma<<<cdiv(EE, TM), blk128, 0, stream>>>(edge_attr, W_ep, b_ep, ea_f, EE, EDC, 1);

    // 3) self-loop edge attrs: mean of incoming ea per dst
    hipMemsetAsync(cnt, 0, sizeof(float) * NN, stream);
    hipMemsetAsync(loop_sum, 0, sizeof(float) * NH, stream);
    count_edges_k<<<cdiv(EE, 256), blk256, 0, stream>>>(dst, cnt, EE);
    sum_loop_ea_k<<<cdiv((long)EE * 32, 128), blk128, 0, stream>>>(ea_f, dst, loop_sum, EE);
    finalize_loop_k<<<cdiv(NH, 256), blk256, 0, stream>>>(loop_sum, cnt, ea_f + (size_t)EE * HIDC, NN);

    // ---------------- layer 1 (heads=4, C=32, concat) ----------------
    gemm_bias_act_wmma<<<cdiv(NN, TM), blk128, 0, stream>>>(h, Wl1, bl1, xl, NN, HIDC, 0);
    gemm_bias_act_wmma<<<cdiv(NN, TM), blk128, 0, stream>>>(h, Wr1, br1, xr, NN, HIDC, 0);
    gemm_bias_act_wmma<<<cdiv(EN, TM), blk128, 0, stream>>>(ea_f, We1, nullptr, ew, (int)EN, HIDC, 0);

    hipMemsetAsync(amax, 0, sizeof(unsigned) * NN * HEADS1, stream);  // enc(-inf) sentinel
    hipMemsetAsync(denom, 0, sizeof(float) * NN * HEADS1, stream);
    hipMemsetAsync(agg, 0, sizeof(float) * NH, stream);

    edge_logits_k<<<cdiv(EN * 32, 128), blk128, 0, stream>>>(
        xl, xr, ew, att1, src, dst, alpha, amax, EE, NN, HEADS1, HIDC / HEADS1);
    edge_softmax_k<<<cdiv(EN * HEADS1, 256), blk256, 0, stream>>>(
        alpha, amax, denom, dst, EE, NN, HEADS1);
    edge_aggregate_k<<<cdiv(EN * 32, 128), blk128, 0, stream>>>(
        alpha, denom, xl, src, dst, agg, EE, NN, HEADS1, HIDC / HEADS1);
    finalize_layer_k<<<cdiv(NH, 256), blk256, 0, stream>>>(agg, bias1, h1, 1, NN);

    // ---------------- layer 2 (heads=1, C=128, mean==identity) ----------------
    gemm_bias_act_wmma<<<cdiv(NN, TM), blk128, 0, stream>>>(h1, Wl2, bl2, xl, NN, HIDC, 0);
    gemm_bias_act_wmma<<<cdiv(NN, TM), blk128, 0, stream>>>(h1, Wr2, br2, xr, NN, HIDC, 0);
    gemm_bias_act_wmma<<<cdiv(EN, TM), blk128, 0, stream>>>(ea_f, We2, nullptr, ew, (int)EN, HIDC, 0);

    hipMemsetAsync(amax, 0, sizeof(unsigned) * NN, stream);
    hipMemsetAsync(denom, 0, sizeof(float) * NN, stream);
    hipMemsetAsync(agg, 0, sizeof(float) * NH, stream);

    edge_logits_k<<<cdiv(EN * 32, 128), blk128, 0, stream>>>(
        xl, xr, ew, att2, src, dst, alpha, amax, EE, NN, 1, OUTC);
    edge_softmax_k<<<cdiv(EN, 256), blk256, 0, stream>>>(
        alpha, amax, denom, dst, EE, NN, 1);
    edge_aggregate_k<<<cdiv(EN * 32, 128), blk128, 0, stream>>>(
        alpha, denom, xl, src, dst, agg, EE, NN, 1, OUTC);
    finalize_layer_k<<<cdiv(NH, 256), blk256, 0, stream>>>(agg, bias2, z_out, 0, NN);
}
